// EdgeEncoder_15453292331280
// MI455X (gfx1250) — matrix operand order repaired and checked by reading, not compiled, not run on hardware
//
#include <hip/hip_runtime.h>

typedef _Float16 f16;
typedef __attribute__((ext_vector_type(16))) _Float16 v16h;
typedef __attribute__((ext_vector_type(8)))  _Float16 v8h;
typedef __attribute__((ext_vector_type(8)))  float    v8f;

#define NN 10000
#define EE 320000
#define HH 4
#define NT_ 8
#define ET_ 22

// ---- order-preserving float<->uint encoding for atomic segment-max ----
static __device__ __forceinline__ unsigned enc_f(float f){
  unsigned u = __float_as_uint(f);
  return (u & 0x80000000u) ? ~u : (u | 0x80000000u);
}
static __device__ __forceinline__ float dec_f(unsigned e){
  unsigned u = (e & 0x80000000u) ? (e & 0x7FFFFFFFu) : ~e;
  return __uint_as_float(u);
}

// ---------------- elementwise / precompute kernels ----------------
__global__ void k_f32_to_f16(const float* __restrict__ in, f16* __restrict__ out, int n){
  int i = blockIdx.x * blockDim.x + threadIdx.x;
  if (i < n) out[i] = (f16)in[i];
}

// f32 [K, Nc] row-major  ->  f16 WMMA-B fragment order:
//   dest[((kt*ntTot + nt)*32 + lane)*16 + elem]
//   lane = (kin/16)*16 + (n%16) ; elem = kin%16 ; kt = k/32 ; nt = n/16
__global__ void k_f32_to_f16_swzB(const float* __restrict__ in, f16* __restrict__ out,
                                  int K, int Nc){
  int i = blockIdx.x * blockDim.x + threadIdx.x;
  if (i >= K * Nc) return;
  int k = i / Nc, n = i % Nc;
  int kt = k >> 5, kin = k & 31;
  int nt = n >> 4, ncol = n & 15;
  int lane = ((kin >> 4) << 4) | ncol;
  int elem = kin & 15;
  size_t d = ((size_t)(kt * (Nc >> 4) + nt) * 32 + lane) * 16 + elem;
  out[d] = (f16)in[i];
}

// etemb [ET, Nc] -> swizzled B with K padded to 32 (rows >= ET_ zero)
__global__ void k_pad_etemb_swzB(const float* __restrict__ et, f16* __restrict__ out, int Nc){
  int i = blockIdx.x * blockDim.x + threadIdx.x;
  if (i >= 32 * Nc) return;
  int k = i / Nc, n = i % Nc;
  int nt = n >> 4, ncol = n & 15;
  int lane = ((k >> 4) << 4) | ncol;
  int elem = k & 15;
  size_t d = ((size_t)nt * 32 + lane) * 16 + elem;
  out[d] = (k < ET_) ? (f16)et[(size_t)k * Nc + n] : (f16)0.0f;
}

__global__ void k_add_nt_f16(const float* __restrict__ x, const float* __restrict__ ntemb,
                             const int* __restrict__ nt, f16* __restrict__ out, int n, int Kd){
  int i = blockIdx.x * blockDim.x + threadIdx.x;
  if (i >= n * Kd) return;
  int node = i / Kd, k = i % Kd;
  out[i] = (f16)(x[i] + ntemb[nt[node] * Kd + k]);
}

__global__ void k_init_maxenc(unsigned* __restrict__ m, int n){
  int i = blockIdx.x * blockDim.x + threadIdx.x;
  if (i < n) m[i] = 0x007FFFFFu;   // enc(-inf)
}

// waedge[k,h] = sum_c Wedge[k, h*C+c] * aedge[h,c]   (Wedge: [64, 4*C])
__global__ void k_waedge(const float* __restrict__ We, const float* __restrict__ ae,
                         float* __restrict__ wa, int C){
  int i = threadIdx.x;
  if (i >= 256) return;
  int k = i >> 2, h = i & 3;
  float s = 0.f;
  for (int c = 0; c < C; ++c) s += We[(size_t)k * 4 * C + h * C + c] * ae[h * C + c];
  wa[i] = s;
}

// etdot[t,h] = sum_c etemb[t, h*C+c] * aedge[h,c]
__global__ void k_etdot(const float* __restrict__ et, const float* __restrict__ ae,
                        float* __restrict__ ed, int C){
  int i = threadIdx.x;
  if (i >= ET_ * 4) return;
  int t = i >> 2, h = i & 3;
  float s = 0.f;
  for (int c = 0; c < C; ++c) s += et[(size_t)t * 4 * C + h * C + c] * ae[h * C + c];
  ed[i] = s;
}

// per-node per-head attention scalars from xh [N, 4*C]
__global__ void k_node_dots(const float* __restrict__ xh, const float* __restrict__ as_,
                            const float* __restrict__ ad_, float* __restrict__ ssrc,
                            float* __restrict__ sdst, int n, int C){
  int i = blockIdx.x * blockDim.x + threadIdx.x;
  if (i >= n * 4) return;
  int node = i >> 2, h = i & 3;
  const float* row = xh + (size_t)node * 4 * C + h * C;
  float a = 0.f, b = 0.f;
  for (int c = 0; c < C; ++c){ float v = row[c]; a += v * as_[h * C + c]; b += v * ad_[h * C + c]; }
  ssrc[i] = a; sdst[i] = b;
}

// ---------------- WMMA f16->f32 GEMM: C[M,Nc] (+)= A[M,K] @ Bswz ----------------
// A: row-major f16 [M, lda]; B pre-swizzled in per-lane fragment order.
// grid.x tiles Nc by 64 (4 x 16-wide subtiles), grid.y tiles M by 16; blockDim = 32 (one wave).
// ntTot = full B width in 16-col tiles; nt0 = starting 16-col tile of this view.
__global__ void k_wmma_gemm(const f16* __restrict__ A, int lda,
                            const f16* __restrict__ Bswz, int ntTot, int nt0,
                            float* __restrict__ C, int ldc,
                            int K, int acc){
  const int lane = threadIdx.x & 31;
  const int mrow = lane & 15;          // A row within tile / C column
  const int grp  = lane >> 4;          // lane group (K/M split per ISA layout)
  const int n0   = blockIdx.x * 64;
  const int m0   = blockIdx.y * 16;
  const f16* Arow = A + (size_t)(m0 + mrow) * lda;

  v8f c[4];
#pragma unroll
  for (int s = 0; s < 4; ++s){
    if (acc){
#pragma unroll
      for (int r = 0; r < 8; ++r)
        c[s][r] = C[(size_t)(m0 + grp * 8 + r) * ldc + n0 + s * 16 + mrow];
    } else {
#pragma unroll
      for (int r = 0; r < 8; ++r) c[s][r] = 0.0f;
    }
  }

  for (int k0 = 0; k0 < K; k0 += 32){
    // A frag: one contiguous 16-half run per lane (k = 16*grp + s, matches swizzled B)
    const v8h a_lo = *reinterpret_cast<const v8h*>(Arow + k0 + grp * 16);
    const v8h a_hi = *reinterpret_cast<const v8h*>(Arow + k0 + grp * 16 + 8);
    const v16h a = __builtin_shufflevector(a_lo, a_hi,
        0, 1, 2, 3, 4, 5, 6, 7, 8, 9, 10, 11, 12, 13, 14, 15);
    const int kt = k0 >> 5;
#pragma unroll
    for (int s = 0; s < 4; ++s){
      // B frag: one contiguous 32-byte load per lane from swizzled storage
      const int nt = nt0 + blockIdx.x * 4 + s;
      const v16h b = *reinterpret_cast<const v16h*>(
          Bswz + ((size_t)(kt * ntTot + nt) * 32 + lane) * 16);
      c[s] = __builtin_amdgcn_wmma_f32_16x16x32_f16(
          false, a, false, b, (short)0, c[s], false, false);
    }
  }

#pragma unroll
  for (int s = 0; s < 4; ++s)
#pragma unroll
    for (int r = 0; r < 8; ++r)
      C[(size_t)(m0 + grp * 8 + r) * ldc + n0 + s * 16 + mrow] = c[s][r];
}

// ---------------- edge-level kernels ----------------
__global__ void k_edge_logits(const float* __restrict__ ea, const int* __restrict__ src,
                              const int* __restrict__ dst, const int* __restrict__ et,
                              const float* __restrict__ ssrc, const float* __restrict__ sdst,
                              const float* __restrict__ wa, const float* __restrict__ ed,
                              float* __restrict__ lg, unsigned* __restrict__ mx, int E){
  int e = blockIdx.x * blockDim.x + threadIdx.x;
  if (e >= E) return;
  int s = src[e], d = dst[e], t = et[e];
  float acc[4];
#pragma unroll
  for (int h = 0; h < 4; ++h) acc[h] = ed[t * 4 + h];
  const float* row = ea + (size_t)e * 64;
  for (int k = 0; k < 64; ++k){
    float v = row[k];
#pragma unroll
    for (int h = 0; h < 4; ++h) acc[h] += v * wa[k * 4 + h];
  }
#pragma unroll
  for (int h = 0; h < 4; ++h){
    float l = acc[h] + ssrc[s * 4 + h] + sdst[d * 4 + h];
    l = (l > 0.0f) ? l : 0.2f * l;                 // leaky_relu(0.2)
    lg[(size_t)e * 4 + h] = l;
    atomicMax(&mx[d * 4 + h], enc_f(l));
  }
}

__global__ void k_softmax_p(const float* __restrict__ lg, const int* __restrict__ dst,
                            const unsigned* __restrict__ mx, float* __restrict__ p,
                            float* __restrict__ den, int E){
  int i = blockIdx.x * blockDim.x + threadIdx.x;
  if (i >= E * 4) return;
  int e = i >> 2, h = i & 3;
  int d = dst[e];
  float m = dec_f(mx[d * 4 + h]);
  float v = __expf(lg[i] - m);
  p[i] = v;
  atomicAdd(&den[d * 4 + h], v);
}

// one wave per (edge, head): scatter alpha*xh[src], alpha*edge_attr, alpha into L2-resident accums
__global__ void k_scatter(const float* __restrict__ p, const float* __restrict__ den,
                          const int* __restrict__ src, const int* __restrict__ dst,
                          const int* __restrict__ et, const float* __restrict__ ea,
                          const float* __restrict__ xh, float* __restrict__ oacc,
                          float* __restrict__ g, float* __restrict__ cntp, int E, int C){
  int gw   = (blockIdx.x * blockDim.x + threadIdx.x) >> 5;
  int lane = threadIdx.x & 31;
  if (gw >= E * 4) return;
  int e = gw >> 2, h = gw & 3;
  int s = src[e], d = dst[e];
  float alpha = p[(size_t)e * 4 + h] / (den[d * 4 + h] + 1e-16f);
  if (lane == 0) atomicAdd(&cntp[(size_t)d * 128 + h * 32 + et[e]], alpha);
  const float* erow = ea + (size_t)e * 64;
  float* grow = g + (size_t)d * 256 + h * 64;
  for (int k = lane; k < 64; k += 32) atomicAdd(&grow[k], alpha * erow[k]);
  const float* xrow = xh + (size_t)s * 4 * C + h * C;
  float* orow = oacc + (size_t)d * 4 * C + h * C;
  for (int c = lane; c < C; c += 32) atomicAdd(&orow[c], alpha * xrow[c]);
}

// ---------------- layer epilogues ----------------
__global__ void k_combine1(const float* __restrict__ oacc, const float* __restrict__ b,
                           float* __restrict__ h1, int total){
  int i = blockIdx.x * blockDim.x + threadIdx.x;
  if (i >= total) return;
  float v = oacc[i] + b[i % 256];
  h1[i] = v > 0.0f ? v : 0.0f;
}

__global__ void k_final(const float* __restrict__ oacc, const float* __restrict__ b2,
                        const float* __restrict__ h1, float* __restrict__ out, int n){
  int i = blockIdx.x * blockDim.x + threadIdx.x;
  if (i >= n * 256) return;
  int node = i / 256, c = i % 256;
  const float* row = oacc + (size_t)node * 1024 + c;
  float s = row[0] + row[256] + row[512] + row[768];
  out[i] = 0.25f * s + b2[c] + h1[i];
}

// ---------------- driver ----------------
extern "C" void kernel_launch(void* const* d_in, const int* in_sizes, int n_in,
                              void* d_out, int out_size, void* d_ws, size_t ws_size,
                              hipStream_t stream){
  (void)in_sizes; (void)n_in; (void)out_size; (void)ws_size;
  const float* x     = (const float*)d_in[0];
  const int*   eidx  = (const int*)  d_in[1];
  const int*   ntype = (const int*)  d_in[2];
  const float* eattr = (const float*)d_in[3];
  const int*   etype = (const int*)  d_in[4];
  const float* l1W   = (const float*)d_in[5];
  const float* l1b   = (const float*)d_in[6];
  const float* l1We  = (const float*)d_in[7];
  const float* l1nt  = (const float*)d_in[8];
  const float* l1et  = (const float*)d_in[9];
  const float* l1as  = (const float*)d_in[10];
  const float* l1ad  = (const float*)d_in[11];
  const float* l1ae  = (const float*)d_in[12];
  const float* l1res = (const float*)d_in[13];
  const float* l2W   = (const float*)d_in[14];
  const float* l2b   = (const float*)d_in[15];
  const float* l2We  = (const float*)d_in[16];
  const float* l2nt  = (const float*)d_in[17];
  const float* l2et  = (const float*)d_in[18];
  const float* l2as  = (const float*)d_in[19];
  const float* l2ad  = (const float*)d_in[20];
  const float* l2ae  = (const float*)d_in[21];
  const int* src = eidx;
  const int* dst = eidx + EE;
  float* out = (float*)d_out;

  char* w = (char*)d_ws; size_t off = 0;
  auto alloc = [&](size_t bytes)->char*{
    char* ptr = w + off; off = (off + bytes + 255) & ~(size_t)255; return ptr;
  };
  float*    xh   = (float*)   alloc((size_t)NN * 1024 * 4);
  float*    oacc = (float*)   alloc((size_t)NN * 1024 * 4);
  float*    h1   = (float*)   alloc((size_t)NN * 256 * 4);
  float*    g    = (float*)   alloc((size_t)NN * 256 * 4);   // [N,H,64] f32 accum
  float*    cntp = (float*)   alloc((size_t)NN * 128 * 4);   // [N,H,32] f32 accum (ET padded)
  f16*      g16  = (f16*)     alloc((size_t)NN * 256 * 2);   // f16 copy of g (A matrix)
  f16*      c16  = (f16*)     alloc((size_t)NN * 128 * 2);   // f16 copy of cntp (A matrix)
  f16*      xa   = (f16*)     alloc((size_t)NN * 256 * 2);
  f16*      xf   = (f16*)     alloc((size_t)NN * 64 * 2);
  float*    pbuf = (float*)   alloc((size_t)EE * 4 * 4);
  float*    lg   = (float*)   alloc((size_t)EE * 4 * 4);
  unsigned* mx   = (unsigned*)alloc((size_t)NN * 4 * 4);
  float*    den  = (float*)   alloc((size_t)NN * 4 * 4);
  float*    ssrc = (float*)   alloc((size_t)NN * 4 * 4);
  float*    sdst = (float*)   alloc((size_t)NN * 4 * 4);
  f16* W1f  = (f16*)alloc(64 * 256 * 2);               // swizzled B
  f16* We1f = (f16*)alloc(64 * 256 * 2);               // swizzled B
  f16* resf = (f16*)alloc(64 * 256 * 2);               // swizzled B
  f16* W2f  = (f16*)alloc((size_t)256 * 1024 * 2);     // swizzled B
  f16* We2f = (f16*)alloc((size_t)64 * 1024 * 2);      // swizzled B
  f16* et1p = (f16*)alloc(32 * 256 * 2);               // swizzled B (K padded to 32)
  f16* et2p = (f16*)alloc(32 * 1024 * 2);              // swizzled B (K padded to 32)
  float* wa1 = (float*)alloc(64 * 4 * 4);
  float* ed1 = (float*)alloc(ET_ * 4 * 4);
  float* wa2 = (float*)alloc(64 * 4 * 4);
  float* ed2 = (float*)alloc(ET_ * 4 * 4);

  auto cvt = [&](const float* s, f16* d, int n){
    k_f32_to_f16<<<(n + 255) / 256, 256, 0, stream>>>(s, d, n);
  };
  auto cvtB = [&](const float* s, f16* d, int K, int Nc){
    k_f32_to_f16_swzB<<<(K * Nc + 255) / 256, 256, 0, stream>>>(s, d, K, Nc);
  };
  cvtB(l1W, W1f, 64, 256); cvtB(l1We, We1f, 64, 256); cvtB(l1res, resf, 64, 256);
  cvtB(l2W, W2f, 256, 1024); cvtB(l2We, We2f, 64, 1024);
  cvt(x, xf, NN * 64);
  k_pad_etemb_swzB<<<(32 * 256 + 255) / 256, 256, 0, stream>>>(l1et, et1p, 256);
  k_pad_etemb_swzB<<<(32 * 1024 + 255) / 256, 256, 0, stream>>>(l2et, et2p, 1024);

  // ===================== layer 1 (Kin=64, C=64, concat) =====================
  (void)hipMemsetAsync(oacc, 0, (size_t)NN * 256 * 4, stream);
  (void)hipMemsetAsync(g,    0, (size_t)NN * 256 * 4, stream);
  (void)hipMemsetAsync(cntp, 0, (size_t)NN * 128 * 4, stream);
  (void)hipMemsetAsync(den,  0, (size_t)NN * 4 * 4, stream);
  k_init_maxenc<<<(NN * 4 + 255) / 256, 256, 0, stream>>>(mx, NN * 4);

  k_add_nt_f16<<<(NN * 64 + 255) / 256, 256, 0, stream>>>(x, l1nt, ntype, xa, NN, 64);
  // xh1 = xa @ W1 : [N,64]x[64,256]
  k_wmma_gemm<<<dim3(4, NN / 16), 32, 0, stream>>>(xa, 64, W1f, 16, 0, xh, 256, 64, 0);
  k_waedge<<<1, 256, 0, stream>>>(l1We, l1ae, wa1, 64);
  k_etdot<<<1, 128, 0, stream>>>(l1et, l1ae, ed1, 64);
  k_node_dots<<<(NN * 4 + 255) / 256, 256, 0, stream>>>(xh, l1as, l1ad, ssrc, sdst, NN, 64);
  k_edge_logits<<<(EE + 255) / 256, 256, 0, stream>>>(eattr, src, dst, etype, ssrc, sdst,
                                                      wa1, ed1, lg, mx, EE);
  k_softmax_p<<<(EE * 4 + 255) / 256, 256, 0, stream>>>(lg, dst, mx, pbuf, den, EE);
  k_scatter<<<(EE * 4 * 32 + 255) / 256, 256, 0, stream>>>(pbuf, den, src, dst, etype,
                                                           eattr, xh, oacc, g, cntp, EE, 64);
  cvt(g, g16, NN * 256);
  cvt(cntp, c16, NN * 128);
  for (int h = 0; h < 4; ++h){
    // oacc[:, h*64:(h+1)*64] += g_h @ Wedge1_h   (B view: tiles h*4..h*4+3 of 16)
    k_wmma_gemm<<<dim3(1, NN / 16), 32, 0, stream>>>(g16 + h * 64, 256, We1f, 16, h * 4,
                                                     oacc + h * 64, 256, 64, 1);
    // oacc[:, h*64:(h+1)*64] += cnt_h @ etemb1_h
    k_wmma_gemm<<<dim3(1, NN / 16), 32, 0, stream>>>(c16 + h * 32, 128, et1p, 16, h * 4,
                                                     oacc + h * 64, 256, 32, 1);
  }
  // residual: oacc += x @ res_proj
  k_wmma_gemm<<<dim3(4, NN / 16), 32, 0, stream>>>(xf, 64, resf, 16, 0, oacc, 256, 64, 1);
  k_combine1<<<(NN * 256 + 255) / 256, 256, 0, stream>>>(oacc, l1b, h1, NN * 256);

  // ===================== layer 2 (Kin=256, C=256, mean) =====================
  (void)hipMemsetAsync(oacc, 0, (size_t)NN * 1024 * 4, stream);
  (void)hipMemsetAsync(g,    0, (size_t)NN * 256 * 4, stream);
  (void)hipMemsetAsync(cntp, 0, (size_t)NN * 128 * 4, stream);
  (void)hipMemsetAsync(den,  0, (size_t)NN * 4 * 4, stream);
  k_init_maxenc<<<(NN * 4 + 255) / 256, 256, 0, stream>>>(mx, NN * 4);

  k_add_nt_f16<<<(NN * 256 + 255) / 256, 256, 0, stream>>>(h1, l2nt, ntype, xa, NN, 256);
  // xh2 = xa @ W2 : [N,256]x[256,1024]
  k_wmma_gemm<<<dim3(16, NN / 16), 32, 0, stream>>>(xa, 256, W2f, 64, 0, xh, 1024, 256, 0);
  k_waedge<<<1, 256, 0, stream>>>(l2We, l2ae, wa2, 256);
  k_etdot<<<1, 128, 0, stream>>>(l2et, l2ae, ed2, 256);
  k_node_dots<<<(NN * 4 + 255) / 256, 256, 0, stream>>>(xh, l2as, l2ad, ssrc, sdst, NN, 256);
  k_edge_logits<<<(EE + 255) / 256, 256, 0, stream>>>(eattr, src, dst, etype, ssrc, sdst,
                                                      wa2, ed2, lg, mx, EE);
  k_softmax_p<<<(EE * 4 + 255) / 256, 256, 0, stream>>>(lg, dst, mx, pbuf, den, EE);
  k_scatter<<<(EE * 4 * 32 + 255) / 256, 256, 0, stream>>>(pbuf, den, src, dst, etype,
                                                           eattr, xh, oacc, g, cntp, EE, 256);
  cvt(g, g16, NN * 256);
  cvt(cntp, c16, NN * 128);
  for (int h = 0; h < 4; ++h){
    // oacc[:, h*256:(h+1)*256] += g_h @ Wedge2_h   (B view: tiles h*16.. of 64)
    k_wmma_gemm<<<dim3(4, NN / 16), 32, 0, stream>>>(g16 + h * 64, 256, We2f, 64, h * 16,
                                                     oacc + h * 256, 1024, 64, 1);
    k_wmma_gemm<<<dim3(4, NN / 16), 32, 0, stream>>>(c16 + h * 32, 128, et2p, 64, h * 16,
                                                     oacc + h * 256, 1024, 32, 1);
  }
  k_final<<<(NN * 256 + 255) / 256, 256, 0, stream>>>(oacc, l2b, h1, out, NN);
}
